// GCN_13846974562486
// MI455X (gfx1250) — matrix operand order repaired and checked by reading, not compiled, not run on hardware
//
#include <hip/hip_runtime.h>
#include <hip/hip_bf16.h>
#include <stdint.h>

#define GN    16384
#define NFEAT 128

typedef __attribute__((ext_vector_type(16))) __bf16 v16bf;
typedef __attribute__((ext_vector_type(8)))  float  v8f;

// -------------------------------------------------------------------------
// Pt[f][n] = sum_k X[n][k] * W[k][f]   (bf16 transposed output, no bias)
// -------------------------------------------------------------------------
template<int KIN, int FOUT>
__global__ __launch_bounds__(256)
void xw_kernel(const float* __restrict__ X,
               const float* __restrict__ W,
               __bf16* __restrict__ Pt)
{
    const int n = blockIdx.x * 256 + threadIdx.x;
    const float* xrow = X + (size_t)n * KIN;

    float acc[FOUT];
#pragma unroll
    for (int f = 0; f < FOUT; ++f) acc[f] = 0.0f;

    for (int k = 0; k < KIN; k += 4) {
        float4 xv = *(const float4*)(xrow + k);
#pragma unroll
        for (int f = 0; f < FOUT; ++f) {
            acc[f] += xv.x * W[(k + 0) * FOUT + f];
            acc[f] += xv.y * W[(k + 1) * FOUT + f];
            acc[f] += xv.z * W[(k + 2) * FOUT + f];
            acc[f] += xv.w * W[(k + 3) * FOUT + f];
        }
    }
#pragma unroll
    for (int f = 0; f < FOUT; ++f)
        Pt[(size_t)f * GN + n] = (__bf16)acc[f];
}

// -------------------------------------------------------------------------
// H[N][F] = relu(adj @ P + bias), P given transposed bf16 Pt[F][N].
// One block = 16 output rows; 4 waves split K; WMMA bf16 with f32 accum.
// -------------------------------------------------------------------------
template<int F>
__global__ __launch_bounds__(128)
void gconv_kernel(const float* __restrict__ adj,
                  const __bf16* __restrict__ Pt,
                  const float* __restrict__ bias,
                  float* __restrict__ H)
{
    constexpr int NT = F / 16;
    __shared__ float red[4][16][F];

    const int strip = blockIdx.x;          // 16-row strip of output
    const int wave  = threadIdx.x >> 5;    // 0..3 (wave32)
    const int lane  = threadIdx.x & 31;
    const int l15   = lane & 15;
    const int hi    = lane >> 4;           // half-wave selector

    v8f acc[NT] = {};

    const int kBeg = wave * (GN / 4);
    const int kEnd = kBeg + (GN / 4);

    const float* aRow = adj + (size_t)(strip * 16 + l15) * GN;

    for (int k0 = kBeg; k0 < kEnd; k0 += 32) {
        // ---- A fragment: 16x32 bf16 from row-major f32 adj ----
        // lanes 0-15: K = k0+[0..15]
        // lanes 16-31: K = k0+[16..31]
        const int kc = k0 + hi * 16;
        float4 f0 = *(const float4*)(aRow + kc);
        float4 f1 = *(const float4*)(aRow + kc + 4);
        float4 f2 = *(const float4*)(aRow + kc + 8);
        float4 f3 = *(const float4*)(aRow + kc + 12);
        v16bf a;
        a[0]  = (__bf16)f0.x; a[1]  = (__bf16)f0.y; a[2]  = (__bf16)f0.z; a[3]  = (__bf16)f0.w;
        a[4]  = (__bf16)f1.x; a[5]  = (__bf16)f1.y; a[6]  = (__bf16)f1.z; a[7]  = (__bf16)f1.w;
        a[8]  = (__bf16)f2.x; a[9]  = (__bf16)f2.y; a[10] = (__bf16)f2.z; a[11] = (__bf16)f2.w;
        a[12] = (__bf16)f3.x; a[13] = (__bf16)f3.y; a[14] = (__bf16)f3.z; a[15] = (__bf16)f3.w;

        // ---- B fragments: lanes 0-15 hold K=k0+[0..15], lanes 16-31 K=k0+[16..31]
        //      for column n = l15 of tile t; contiguous in Pt rows ----
#pragma unroll
        for (int t = 0; t < NT; ++t) {
            const __bf16* bp = Pt + (size_t)(t * 16 + l15) * GN + k0 + hi * 16;
            v16bf b = *(const v16bf*)bp;   // 32B, 32B-aligned -> 2x b128
            acc[t] = __builtin_amdgcn_wmma_f32_16x16x32_bf16(
                false, a, false, b, (short)0, acc[t], false, false);
        }
    }

    // ---- dump C fragments (M = g + hi*8, N = t*16 + l15) to LDS ----
#pragma unroll
    for (int t = 0; t < NT; ++t)
#pragma unroll
        for (int g = 0; g < 8; ++g)
            red[wave][g + hi * 8][t * 16 + l15] = acc[t][g];
    __syncthreads();

    // ---- cross-wave K reduction + bias + relu + store ----
    for (int idx = threadIdx.x; idx < 16 * F; idx += 128) {
        const int m = idx / F;
        const int f = idx % F;
        float s = red[0][m][f] + red[1][m][f] + red[2][m][f] + red[3][m][f];
        s = fmaxf(s + bias[f], 0.0f);
        H[(size_t)(strip * 16 + m) * F + f] = s;
    }
}

// -------------------------------------------------------------------------
// Stage-1 column sums of H3[N][64] (deterministic, no float atomics)
// -------------------------------------------------------------------------
__global__ __launch_bounds__(256)
void colpart_kernel(const float* __restrict__ H, float* __restrict__ partials)
{
    __shared__ float sm[256];
    const int f  = threadIdx.x & 63;
    const int rg = threadIdx.x >> 6;      // 0..3
    const int r0 = blockIdx.x * 128;
    float s = 0.0f;
    for (int j = 0; j < 128; j += 4)
        s += H[(size_t)(r0 + rg + j) * 64 + f];
    sm[threadIdx.x] = s;
    __syncthreads();
    if (threadIdx.x < 64)
        partials[blockIdx.x * 64 + f] = sm[f] + sm[64 + f] + sm[128 + f] + sm[192 + f];
}

// -------------------------------------------------------------------------
// mean -> fc1+relu -> fc2 -> softmax  (single block, 64 threads)
// -------------------------------------------------------------------------
__global__ void head_kernel(const float* __restrict__ partials,
                            const float* __restrict__ fcW1,
                            const float* __restrict__ fcb1,
                            const float* __restrict__ fcW2,
                            const float* __restrict__ fcb2,
                            float* __restrict__ out)
{
    __shared__ float y[64];
    __shared__ float z[32];
    __shared__ float sl[2];
    const int t = threadIdx.x;            // 64 threads

    float s = 0.0f;
    for (int b = 0; b < 128; ++b) s += partials[b * 64 + t];
    y[t] = s * (1.0f / (float)GN);
    __syncthreads();

    if (t < 32) {
        float a = fcb1[t];
        for (int f = 0; f < 64; ++f) a += y[f] * fcW1[f * 32 + t];
        z[t] = fmaxf(a, 0.0f);
    }
    __syncthreads();

    if (t < 2) {
        float a = fcb2[t];
        for (int j = 0; j < 32; ++j) a += z[j] * fcW2[j * 2 + t];
        sl[t] = a;
    }
    __syncthreads();

    if (t == 0) {
        float m  = fmaxf(sl[0], sl[1]);
        float e0 = __expf(sl[0] - m);
        float e1 = __expf(sl[1] - m);
        float inv = 1.0f / (e0 + e1);
        out[0] = e0 * inv;
        out[1] = e1 * inv;
    }
}

// -------------------------------------------------------------------------
extern "C" void kernel_launch(void* const* d_in, const int* in_sizes, int n_in,
                              void* d_out, int out_size, void* d_ws, size_t ws_size,
                              hipStream_t stream)
{
    const float* x    = (const float*)d_in[0];
    const float* adj  = (const float*)d_in[1];
    // d_in[2] = idx_map (unused by the reference)
    const float* W1   = (const float*)d_in[3];
    const float* b1   = (const float*)d_in[4];
    const float* W2   = (const float*)d_in[5];
    const float* b2   = (const float*)d_in[6];
    const float* W3   = (const float*)d_in[7];
    const float* b3   = (const float*)d_in[8];
    const float* fcW1 = (const float*)d_in[9];
    const float* fcb1 = (const float*)d_in[10];
    const float* fcW2 = (const float*)d_in[11];
    const float* fcb2 = (const float*)d_in[12];
    float* out = (float*)d_out;

    char* ws = (char*)d_ws;
    __bf16* Pt    = (__bf16*)(ws);                                   // 64*GN*2  = 2 MB
    float*  H     = (float*)(ws + (size_t)64 * GN * 2);              // GN*64*4  = 4 MB
    float*  parts = (float*)(ws + (size_t)64 * GN * 2
                                + (size_t)GN * 64 * 4);              // 128*64*4 = 32 KB

    // Layer 1: P1 = x @ W1 ; H1 = relu(adj @ P1 + b1)
    xw_kernel<NFEAT, 32><<<GN / 256, 256, 0, stream>>>(x, W1, Pt);
    gconv_kernel<32><<<GN / 16, 128, 0, stream>>>(adj, Pt, b1, H);
    // Layer 2
    xw_kernel<32, 48><<<GN / 256, 256, 0, stream>>>(H, W2, Pt);
    gconv_kernel<48><<<GN / 16, 128, 0, stream>>>(adj, Pt, b2, H);
    // Layer 3
    xw_kernel<48, 64><<<GN / 256, 256, 0, stream>>>(H, W3, Pt);
    gconv_kernel<64><<<GN / 16, 128, 0, stream>>>(adj, Pt, b3, H);
    // Readout + MLP head + softmax
    colpart_kernel<<<128, 256, 0, stream>>>(H, parts);
    head_kernel<<<1, 64, 0, stream>>>(parts, fcW1, fcb1, fcW2, fcb2, out);
}
